// MinGRUCell_84653805404386
// MI455X (gfx1250) — matrix operand order repaired and checked by reading, not compiled, not run on hardware
//
#include <hip/hip_runtime.h>

// ---------------------------------------------------------------------------
// MinGRU on MI455X (gfx1250, wave32):
//   z  = sigmoid((x @ Wz^T + bz) * inv_tau),  h~ = x @ Wh^T + bh
//   h_t = (1-z_t) h_{t-1} + z_t h~_t     (sequential over T)
//
// GEMMs are fp32-compute-bound, so run them as bf16x3 emulation on
// V_WMMA_F32_16X16X32_BF16 (K=32/instr vs K=4 for the fp32 WMMA):
//   A*B ~= Ahi*Bhi + Ahi*Blo + Alo*Bhi, fp32 accumulate.
// The hi/lo split is done in registers while staging tiles into LDS.
// Wave tile 64x32 (acc[4][2]) -> 24 WMMAs per 24 ds_load_b128 per K=32.
// ---------------------------------------------------------------------------

typedef __attribute__((ext_vector_type(16))) __bf16          v16bf;
typedef __attribute__((ext_vector_type(8)))  float           v8f;
typedef __attribute__((ext_vector_type(8)))  unsigned short  v8u;
typedef __attribute__((ext_vector_type(4)))  unsigned short  v4u;

union Frag {                 // 32 bytes: one WMMA bf16 A/B fragment
    v16bf bf;
    v8u   u[2];
};

#define BB   32
#define TT   2048
#define DD   512            // D_IN == K
#define HH   512            // H
#define MM   (BB * TT)      // 65536 GEMM rows

#define MTILE 256           // rows per workgroup
#define NTILE 64            // cols per workgroup (within one weight matrix)
#define KTILE 32            // K per LDS stage == WMMA K
#define LDE   40            // LDS row stride in bf16 (80 B): 16B-aligned frag
                            // reads, conflict-free 20-dword stride

// float -> bf16 (RNE) and back, as raw ushort bits
__device__ __forceinline__ unsigned short f2bf(float f) {
    unsigned int u = __float_as_uint(f);
    u += 0x7FFFu + ((u >> 16) & 1u);
    return (unsigned short)(u >> 16);
}
__device__ __forceinline__ float bf2f(unsigned short h) {
    return __uint_as_float(((unsigned int)h) << 16);
}

// ---------------------------------------------------------------------------
// Kernel 1: invtau[m] = 1 / (1 + softplus(alpha)*sigmoid(mw*mmag[m] + mb))
// (reciprocal precomputed here so the GEMM epilogue multiplies, not divides)
// ---------------------------------------------------------------------------
__global__ void mingru_tau_kernel(const float* __restrict__ mmag,
                                  const float* __restrict__ mw,
                                  const float* __restrict__ mb,
                                  const float* __restrict__ alpha,
                                  float* __restrict__ invtau) {
    int i = blockIdx.x * blockDim.x + threadIdx.x;
    if (i < MM) {
        float a = log1pf(__expf(alpha[0]));                        // softplus
        float s = 1.0f / (1.0f + __expf(-(mw[0] * mmag[i] + mb[0])));
        invtau[i] = 1.0f / (1.0f + a * s);
    }
}

// ---------------------------------------------------------------------------
// Kernel 2: bf16x3 WMMA GEMM + fused epilogue.
//   blockIdx.x 0..7  -> Wz path: sigmoid((acc+bz)*invtau) -> zbuf (ws)
//   blockIdx.x 8..15 -> Wh path: (acc+bh)                 -> out
//   blockIdx.y       -> M block (256 rows)
// 256 threads = 8 waves in a 4(M) x 2(N) grid; each wave owns a 64x32 tile
// as 4x2 v8f accumulators -> 24 v_wmma_f32_16x16x32_bf16 per K=32 step.
// ---------------------------------------------------------------------------
__global__ __launch_bounds__(256)
void mingru_gemm_kernel(const float* __restrict__ x,
                        const float* __restrict__ Wz,
                        const float* __restrict__ bz,
                        const float* __restrict__ Wh,
                        const float* __restrict__ bh,
                        const float* __restrict__ invtau,
                        float* __restrict__ zbuf,
                        float* __restrict__ out) {
    __shared__ __align__(16) unsigned short lAh[MTILE * LDE];   // 20 KB
    __shared__ __align__(16) unsigned short lAl[MTILE * LDE];   // 20 KB
    __shared__ __align__(16) unsigned short lBh[NTILE * LDE];   //  5 KB
    __shared__ __align__(16) unsigned short lBl[NTILE * LDE];   //  5 KB

    const bool is_z  = (blockIdx.x < 8);
    const int  ncol0 = (blockIdx.x & 7) * NTILE;     // col base within W
    const int  mbase = blockIdx.y * MTILE;           // global row base

    const float* __restrict__ W = is_z ? Wz : Wh;

    const int tid  = threadIdx.x;
    const int lane = tid & 31;
    const int wave = tid >> 5;
    const int wm   = wave & 3;                       // M wave coord (0..3)
    const int wn   = wave >> 2;                      // N wave coord (0..1)
    const int lm   = lane & 15;                      // row/col within frag
    const int lhi  = lane >> 4;                      // half select (0/1)

    v8f acc[4][2];
    #pragma unroll
    for (int mi = 0; mi < 4; ++mi) {
        acc[mi][0] = (v8f)0.0f;
        acc[mi][1] = (v8f)0.0f;
    }

    const int aOff0 = 8 * lhi;        // A run 1: K in [8h, 8h+8)
    const int aOff1 = 16 + 8 * lhi;   // A run 2: K in [16+8h, 16+8h+8)
    const int bOff  = 16 * lhi;       // unused; B frags use aOff0/aOff1 (same K as A)

    for (int k0 = 0; k0 < DD; k0 += KTILE) {
        __syncthreads();
        // ---- stage A: 256 x 32 fp32 -> bf16 hi/lo (2048 float4; 8/thread)
        #pragma unroll
        for (int i = 0; i < 8; ++i) {
            int idx = tid + i * 256;
            int row = idx >> 3;                      // 8 float4 per row
            int q   = idx & 7;
            float4 v = *(const float4*)&x[(size_t)(mbase + row) * DD + k0 + q * 4];
            v4u h, l;
            h.x = f2bf(v.x); l.x = f2bf(v.x - bf2f(h.x));
            h.y = f2bf(v.y); l.y = f2bf(v.y - bf2f(h.y));
            h.z = f2bf(v.z); l.z = f2bf(v.z - bf2f(h.z));
            h.w = f2bf(v.w); l.w = f2bf(v.w - bf2f(h.w));
            *(v4u*)&lAh[row * LDE + q * 4] = h;
            *(v4u*)&lAl[row * LDE + q * 4] = l;
        }
        // ---- stage B: 64 x 32 fp32 -> bf16 hi/lo (512 float4; 2/thread)
        #pragma unroll
        for (int i = 0; i < 2; ++i) {
            int idx = tid + i * 256;
            int row = idx >> 3;
            int q   = idx & 7;
            float4 v = *(const float4*)&W[(size_t)(ncol0 + row) * DD + k0 + q * 4];
            v4u h, l;
            h.x = f2bf(v.x); l.x = f2bf(v.x - bf2f(h.x));
            h.y = f2bf(v.y); l.y = f2bf(v.y - bf2f(h.y));
            h.z = f2bf(v.z); l.z = f2bf(v.z - bf2f(h.z));
            h.w = f2bf(v.w); l.w = f2bf(v.w - bf2f(h.w));
            *(v4u*)&lBh[row * LDE + q * 4] = h;
            *(v4u*)&lBl[row * LDE + q * 4] = l;
        }
        __syncthreads();

        // ---- fragment loads (ISA 16-bit A/B layouts) ----
        Frag fAh[4], fAl[4], fBh[2], fBl[2];
        #pragma unroll
        for (int mi = 0; mi < 4; ++mi) {
            const unsigned short* rh = &lAh[(wm * 64 + mi * 16 + lm) * LDE];
            const unsigned short* rl = &lAl[(wm * 64 + mi * 16 + lm) * LDE];
            fAh[mi].u[0] = *(const v8u*)(rh + aOff0);
            fAh[mi].u[1] = *(const v8u*)(rh + aOff1);
            fAl[mi].u[0] = *(const v8u*)(rl + aOff0);
            fAl[mi].u[1] = *(const v8u*)(rl + aOff1);
        }
        #pragma unroll
        for (int ni = 0; ni < 2; ++ni) {
            const unsigned short* rh = &lBh[(wn * 32 + ni * 16 + lm) * LDE];
            const unsigned short* rl = &lBl[(wn * 32 + ni * 16 + lm) * LDE];
            fBh[ni].u[0] = *(const v8u*)(rh + aOff0);
            fBh[ni].u[1] = *(const v8u*)(rh + aOff1);
            fBl[ni].u[0] = *(const v8u*)(rl + aOff0);
            fBl[ni].u[1] = *(const v8u*)(rl + aOff1);
        }

        // ---- 24 WMMAs: hi*hi + hi*lo + lo*hi per 16x16 subtile ----
        #pragma unroll
        for (int mi = 0; mi < 4; ++mi) {
            #pragma unroll
            for (int ni = 0; ni < 2; ++ni) {
                acc[mi][ni] = __builtin_amdgcn_wmma_f32_16x16x32_bf16(
                    false, fAh[mi].bf, false, fBh[ni].bf, (short)0,
                    acc[mi][ni], false, false);
                acc[mi][ni] = __builtin_amdgcn_wmma_f32_16x16x32_bf16(
                    false, fAh[mi].bf, false, fBl[ni].bf, (short)0,
                    acc[mi][ni], false, false);
                acc[mi][ni] = __builtin_amdgcn_wmma_f32_16x16x32_bf16(
                    false, fAl[mi].bf, false, fBh[ni].bf, (short)0,
                    acc[mi][ni], false, false);
            }
        }
    }

    // ---- epilogue: C/D layout is row = r + 8*lhi, col = lane%16 ----
    #pragma unroll
    for (int mi = 0; mi < 4; ++mi) {
        #pragma unroll
        for (int ni = 0; ni < 2; ++ni) {
            const int n = ncol0 + wn * 32 + ni * 16 + lm;      // 0..511 in W
            const float bias = is_z ? bz[n] : bh[n];
            #pragma unroll
            for (int r = 0; r < 8; ++r) {
                const int   m = mbase + wm * 64 + mi * 16 + 8 * lhi + r;
                const float c = acc[mi][ni][r] + bias;
                if (is_z) {
                    const float it = invtau[m];
                    zbuf[(size_t)m * HH + n] =
                        1.0f / (1.0f + __expf(-(c * it)));
                } else {
                    out[(size_t)m * HH + n] = c;
                }
            }
        }
    }
}

// ---------------------------------------------------------------------------
// Kernel 3: sequential recurrence, one thread per (b, h) channel.
// out[] initially holds h~; updated in place. Unroll x4 so the 8 loads of a
// group batch ahead of the dependent FMA chain.
// ---------------------------------------------------------------------------
__global__ __launch_bounds__(256)
void mingru_scan_kernel(const float* __restrict__ zbuf,
                        float* __restrict__ out) {
    const int tid = blockIdx.x * blockDim.x + threadIdx.x;   // 0..16383
    const int b   = tid >> 9;                                // /512
    const int hh  = tid & (HH - 1);
    size_t base = (size_t)b * TT * HH + hh;
    float h = 0.0f;
    for (int t = 0; t < TT; t += 4) {
        const size_t i0 = base + (size_t)t * HH;
        float z0 = zbuf[i0];
        float z1 = zbuf[i0 + HH];
        float z2 = zbuf[i0 + 2 * HH];
        float z3 = zbuf[i0 + 3 * HH];
        float g0 = out[i0];
        float g1 = out[i0 + HH];
        float g2 = out[i0 + 2 * HH];
        float g3 = out[i0 + 3 * HH];
        h = h + z0 * (g0 - h); out[i0]          = h;
        h = h + z1 * (g1 - h); out[i0 + HH]     = h;
        h = h + z2 * (g2 - h); out[i0 + 2 * HH] = h;
        h = h + z3 * (g3 - h); out[i0 + 3 * HH] = h;
    }
}

// ---------------------------------------------------------------------------
// Launch
// ---------------------------------------------------------------------------
extern "C" void kernel_launch(void* const* d_in, const int* in_sizes, int n_in,
                              void* d_out, int out_size, void* d_ws, size_t ws_size,
                              hipStream_t stream) {
    const float* x     = (const float*)d_in[0];
    const float* mmag  = (const float*)d_in[1];
    const float* Wz    = (const float*)d_in[2];
    const float* bz    = (const float*)d_in[3];
    const float* Wh    = (const float*)d_in[4];
    const float* bh    = (const float*)d_in[5];
    const float* mw    = (const float*)d_in[6];
    const float* mb    = (const float*)d_in[7];
    const float* alpha = (const float*)d_in[8];
    float* out = (float*)d_out;

    // workspace: [ zbuf : M*H floats | invtau : M floats ]
    float* zbuf   = (float*)d_ws;
    float* invtau = zbuf + (size_t)MM * HH;

    mingru_tau_kernel<<<MM / 256, 256, 0, stream>>>(mmag, mw, mb, alpha,
                                                    invtau);

    dim3 grid(16, MM / MTILE);  // 16 N-blocks (8 z + 8 h), 256 M-blocks
    mingru_gemm_kernel<<<grid, 256, 0, stream>>>(x, Wz, bz, Wh, bh, invtau,
                                                 zbuf, out);

    mingru_scan_kernel<<<(BB * HH) / 256, 256, 0, stream>>>(zbuf, out);
}
